// TPHunyuanImage3Attention_65060164600409
// MI455X (gfx1250) — matrix operand order repaired and checked by reading, not compiled, not run on hardware
//
#include <hip/hip_runtime.h>
#include <hip/hip_bf16.h>
#include <math.h>

// Problem constants (from reference setup_inputs)
#define BATCH 2
#define SEQ   2048
#define HID   4096
#define NH    32
#define NKV   8
#define HD    128
#define QKVN  (HID + 2 * NKV * HD)   // 6144
#define NTOK  (BATCH * SEQ)          // 4096

typedef __bf16 bf16;
typedef __attribute__((ext_vector_type(16))) __bf16 v16bf;
typedef __attribute__((ext_vector_type(8)))  __bf16 v8bf;
typedef __attribute__((ext_vector_type(8)))  float  v8f;

// ---------------------------------------------------------------------------
// CDNA5 async copy: global -> LDS, 16 bytes per lane, tracked by ASYNCcnt.
// VDST VGPR holds the wave-relative LDS byte address (added to the LDS base
// and the instruction offset).  Generic pointers to LDS keep the byte offset in
// addr[31:0] (ISA 10.2 aperture mapping), so truncation yields the LDS addr.
// ---------------------------------------------------------------------------
__device__ __forceinline__ void async_load_b128(void* lds, const void* g) {
    unsigned l = (unsigned)(size_t)lds;
    asm volatile("global_load_async_to_lds_b128 %0, %1, off"
                 :: "v"(l), "v"(g) : "memory");
}

#if __has_builtin(__builtin_amdgcn_s_wait_asynccnt)
#define WAIT_ASYNC(n) __builtin_amdgcn_s_wait_asynccnt(n)
#else
#define WAIT_ASYNC(n) asm volatile("s_wait_asynccnt %0" :: "i"(n) : "memory")
#endif

__device__ __forceinline__ void wmma_bf16(v8f& acc, v16bf a, v16bf b) {
    // D = A(16x32 bf16) * B(32x16 bf16) + C(16x16 f32)
    acc = __builtin_amdgcn_wmma_f32_16x16x32_bf16(false, a, false, b,
                                                  (short)0, acc, false, false);
}

// A fragment: 16x32 tile from row-major bf16 source, leading dim `ld` elements.
// ISA 16-bit A layout: lane m = l&15, half = l>>4; elems 0..7 -> k = 8*half+0..7,
// elems 8..15 -> k = 16 + 8*half + 0..7.  Two contiguous 16B loads.
__device__ __forceinline__ v16bf load_frag_a(const bf16* base, int ld) {
    int lane = threadIdx.x & 31;
    int m = lane & 15, hf = lane >> 4;
    const bf16* p = base + m * ld + 8 * hf;
    union { v16bf v; v8bf h[2]; } u;
    u.h[0] = *(const v8bf*)p;
    u.h[1] = *(const v8bf*)(p + 16);
    return u.v;
}

// B fragment where the source holds B^T, i.e. src[n][k] row-major ([N,K]).
// ISA 16-bit B layout: lane n = l&15, half = l>>4; elems 0..7 -> k = 8*half+0..7,
// elems 8..15 -> k = 16 + 8*half + 0..7.
__device__ __forceinline__ v16bf load_frag_bt(const bf16* base, int ld) {
    int lane = threadIdx.x & 31;
    int n = lane & 15, hf = lane >> 4;
    const bf16* p = base + n * ld + 8 * hf;
    union { v16bf v; v8bf h[2]; } u;
    u.h[0] = *(const v8bf*)p;
    u.h[1] = *(const v8bf*)(p + 16);
    return u.v;
}

// ---------------------------------------------------------------------------
// One-shot fp32 -> bf16 cast (weights/activations), 8 elements per thread.
// ---------------------------------------------------------------------------
__global__ __launch_bounds__(256) void cast_f32_bf16(
    const float* __restrict__ src, bf16* __restrict__ dst, int n8) {
    int i = blockIdx.x * 256 + threadIdx.x;
    if (i >= n8) return;
    size_t base = (size_t)i * 8;
    float4 a = *(const float4*)(src + base);
    float4 b = *(const float4*)(src + base + 4);
    union { uint4 u; bf16 e[8]; } t;
    t.e[0] = (bf16)a.x; t.e[1] = (bf16)a.y; t.e[2] = (bf16)a.z; t.e[3] = (bf16)a.w;
    t.e[4] = (bf16)b.x; t.e[5] = (bf16)b.y; t.e[6] = (bf16)b.z; t.e[7] = (bf16)b.w;
    *(uint4*)(dst + base) = t.u;
}

// ---------------------------------------------------------------------------
// All-bf16 tiled GEMM: C[M,N] (f32) = A[M,K] * B[N,K]^T.
// Block tile 128x128, BK=32, double-buffered LDS fed by async-to-LDS copies.
// 8 waves: wave (wm 0..3, wn 0..1) owns rows 32*wm..+31, cols 64*wn..+63.
// ---------------------------------------------------------------------------
__global__ __launch_bounds__(256) void gemm_wmma(const bf16* __restrict__ A,
                                                 const bf16* __restrict__ B,
                                                 float* __restrict__ C,
                                                 int M, int N, int K) {
    const int LDT = 40;                 // 32 + 8 pad: 16B-aligned rows, no bank walls
    __shared__ bf16 As[2][128 * 40];    // 2 x 10 KB
    __shared__ bf16 Bs[2][128 * 40];    // 2 x 10 KB

    const int tid  = threadIdx.x;
    const int lane = tid & 31;
    const int wid  = tid >> 5;
    const int wm   = wid & 3;
    const int wn   = wid >> 2;
    const int bm   = blockIdx.y;
    const int bn   = blockIdx.x;

    v8f acc[2][4];
    for (int i = 0; i < 2; ++i)
        for (int j = 0; j < 4; ++j)
            for (int r = 0; r < 8; ++r) acc[i][j][r] = 0.f;

    // Issue async copies of one 128x32 tile of A and of B into buffer `buf`.
    auto issue_tiles = [&](int buf, int k0) {
        for (int i = 0; i < 2; ++i) {
            int idx = tid + i * 256;        // 512 16B chunks per operand
            int r   = idx >> 2;             // 4 chunks per 32-elem row
            int c   = (idx & 3) * 8;
            async_load_b128(&As[buf][r * LDT + c],
                            A + (size_t)(bm * 128 + r) * K + k0 + c);
            async_load_b128(&Bs[buf][r * LDT + c],
                            B + (size_t)(bn * 128 + r) * K + k0 + c);
        }
    };

    const int nk = K / 32;
    issue_tiles(0, 0);

    for (int kk = 0; kk < nk; ++kk) {
        const int buf = kk & 1;
        WAIT_ASYNC(0);                 // this wave's async copies have landed
        __syncthreads();               // whole tile published; prev buf free
        if (kk + 1 < nk) issue_tiles(buf ^ 1, (kk + 1) * 32);

        v16bf af0 = load_frag_a(&As[buf][(wm * 32 + 0)  * LDT], LDT);
        v16bf af1 = load_frag_a(&As[buf][(wm * 32 + 16) * LDT], LDT);
        for (int j = 0; j < 4; ++j) {
            v16bf bb = load_frag_bt(&Bs[buf][(wn * 64 + j * 16) * LDT], LDT);
            wmma_bf16(acc[0][j], af0, bb);
            wmma_bf16(acc[1][j], af1, bb);
        }
        __syncthreads();               // done reading buf before it is refilled
    }

    // Write back: C/D layout -> lane n = l&15, half = l>>4, vgpr r -> m = r+8*half
    const int n_ = lane & 15;
    const int hf = lane >> 4;
    for (int i = 0; i < 2; ++i)
        for (int j = 0; j < 4; ++j)
            for (int r = 0; r < 8; ++r) {
                int row = bm * 128 + wm * 32 + i * 16 + r + 8 * hf;
                int col = bn * 128 + wn * 64 + j * 16 + n_;
                C[(size_t)row * N + col] = acc[i][j][r];
            }
}

// ---------------------------------------------------------------------------
// RoPE + RMSNorm + bf16 cast. One block per token, one wave per head-task.
// Tasks: 0..31 = Q heads, 32..39 = K heads, 40..47 = V heads (plain cast).
// Lane owns d, d+32, d+64, d+96 so rotate_half is lane-local.
// ---------------------------------------------------------------------------
__global__ __launch_bounds__(256) void rope_norm_kernel(
    const float* __restrict__ qkv, const float* __restrict__ cosp,
    const float* __restrict__ sinp, const float* __restrict__ qw,
    const float* __restrict__ kw, bf16* __restrict__ qg,
    bf16* __restrict__ kg, bf16* __restrict__ vg) {
    const int t    = blockIdx.x;          // token index in [0, NTOK)
    const int b    = t / SEQ;
    const int s    = t - b * SEQ;
    const int lane = threadIdx.x & 31;
    const int wid  = threadIdx.x >> 5;

    const float* row = qkv + (size_t)t * QKVN;
    const float* cr  = cosp + (size_t)t * HD;
    const float* sr  = sinp + (size_t)t * HD;

    const int d0 = lane, d1 = lane + 32, d2 = lane + 64, d3 = lane + 96;

    for (int task = wid; task < 48; task += 8) {
        const float* src;
        if (task < 32)      src = row + task * HD;
        else if (task < 40) src = row + HID + (task - 32) * HD;
        else                src = row + HID + NKV * HD + (task - 40) * HD;

        float x0 = src[d0], x1 = src[d1], x2 = src[d2], x3 = src[d3];
        float y0, y1, y2, y3;

        if (task < 40) {
            // RoPE: y[i] = x[i]*cos[i] + rot(x)[i]*sin[i]
            float r0 = -x2, r1 = -x3, r2 = x0, r3 = x1;
            y0 = x0 * cr[d0] + r0 * sr[d0];
            y1 = x1 * cr[d1] + r1 * sr[d1];
            y2 = x2 * cr[d2] + r2 * sr[d2];
            y3 = x3 * cr[d3] + r3 * sr[d3];
            // RMSNorm over D=128 (wave32 reduction, 4 elems per lane)
            float ss = y0 * y0 + y1 * y1 + y2 * y2 + y3 * y3;
            for (int off = 16; off > 0; off >>= 1)
                ss += __shfl_xor(ss, off, 32);
            float sc = rsqrtf(ss * (1.0f / HD) + 1e-6f);
            const float* w = (task < 32) ? qw : kw;
            y0 *= sc * w[d0]; y1 *= sc * w[d1];
            y2 *= sc * w[d2]; y3 *= sc * w[d3];
        } else {
            y0 = x0; y1 = x1; y2 = x2; y3 = x3;
        }

        bf16* dst;
        if (task < 32)      dst = qg + ((size_t)(b * NH  + task)        * SEQ + s) * HD;
        else if (task < 40) dst = kg + ((size_t)(b * NKV + (task - 32)) * SEQ + s) * HD;
        else                dst = vg + ((size_t)(b * NKV + (task - 40)) * SEQ + s) * HD;
        dst[d0] = (bf16)y0; dst[d1] = (bf16)y1;
        dst[d2] = (bf16)y2; dst[d3] = (bf16)y3;
    }
}

// ---------------------------------------------------------------------------
// Flash attention: block = 64 query rows x full head dim for one (b,h).
// Key tiles of 32, online softmax (cross-wave via LDS), WMMA for QK^T and PV.
// Q/K tiles arrive via async-to-LDS; V is transposed manually into LDS.
// ---------------------------------------------------------------------------
__global__ __launch_bounds__(256) void flash_attn_kernel(
    const bf16* __restrict__ qg, const bf16* __restrict__ kg,
    const bf16* __restrict__ vg, bf16* __restrict__ og) {
    const int LQ = 136;   // Q/K row pad (128+8)
    const int LV = 40;    // Vt row pad (32+8)
    const int LS = 36;    // score row pad (32+4)
    const int LP = 40;    // P row pad (32+8)
    __shared__ bf16  Qs[64 * 136];    // 17408 B
    __shared__ bf16  Ks[32 * 136];    //  8704 B
    __shared__ bf16  Vt[128 * 40];    // 10240 B  (transposed: Vt[d][kv])
    __shared__ float Sc[64 * 36];     //  9216 B
    __shared__ bf16  Ps[64 * 40];     //  5120 B
    __shared__ float mrow[64], lrow[64], arow[64];

    const int tid  = threadIdx.x;
    const int lane = tid & 31;
    const int wid  = tid >> 5;
    const int q0   = blockIdx.x * 64;
    const int bh   = blockIdx.y;
    const int b    = bh / NH;
    const int h    = bh - b * NH;
    const int kh   = h / (NH / NKV);

    const bf16* Qg = qg + ((size_t)(b * NH  + h)  * SEQ + q0) * HD;
    const bf16* Kg = kg +  (size_t)(b * NKV + kh) * SEQ * HD;
    const bf16* Vg = vg +  (size_t)(b * NKV + kh) * SEQ * HD;

    // Async-load Q tile (64x128 bf16) once; completion covered by the first
    // WAIT_ASYNC inside the loop before the compute barrier.
    for (int i = 0; i < 4; ++i) {
        int idx = tid + i * 256;            // 1024 16B chunks
        int r = idx >> 4, c = (idx & 15) * 8;
        async_load_b128(&Qs[r * LQ + c], Qg + (size_t)r * HD + c);
    }
    if (tid < 64) { mrow[tid] = -INFINITY; lrow[tid] = 0.f; arow[tid] = 0.f; }

    const int mi = wid >> 1;         // 0..3: 16-row tile
    const int sn = wid & 1;          // score col tile (0..1), 32 keys
    const int nb = (wid & 1) * 4;    // PV col tile base (0 or 4) of 8

    v8f accO[4];
    for (int j = 0; j < 4; ++j)
        for (int r = 0; r < 8; ++r) accO[j][r] = 0.f;

    const float scale = 0.0883883476483184f;   // 1/sqrt(128)
    const int n_ = lane & 15;
    const int hf = lane >> 4;

    for (int kt = 0; kt < SEQ / 32; ++kt) {
        __syncthreads();   // previous iteration's LDS reads done before reload
        // Stage K tile (32x128, async) and V tile transposed (Vt[d][kv]).
        for (int i = 0; i < 2; ++i) {
            int idx = tid + i * 256;        // 512 16B chunks
            int r = idx >> 4, c = (idx & 15) * 8;
            size_t grow = (size_t)(kt * 32 + r) * HD + c;
            async_load_b128(&Ks[r * LQ + c], Kg + grow);
            union { uint4 u; bf16 e[8]; } vv;
            vv.u = *(const uint4*)(Vg + grow);
            for (int e = 0; e < 8; ++e) Vt[(c + e) * LV + r] = vv.e[e];
        }
        WAIT_ASYNC(0);                 // async K (and first-iter Q) landed
        __syncthreads();

        // S = Q * K^T : each wave one 16x16 tile (mi, sn), K-dim 128 = 4 chunks.
        {
            v8f s;
            for (int r = 0; r < 8; ++r) s[r] = 0.f;
            for (int c = 0; c < 4; ++c) {
                v16bf qa = load_frag_a (&Qs[(mi * 16) * LQ + c * 32], LQ);
                v16bf kb = load_frag_bt(&Ks[(sn * 16) * LQ + c * 32], LQ);
                wmma_bf16(s, qa, kb);
            }
            for (int r = 0; r < 8; ++r)
                Sc[(mi * 16 + r + 8 * hf) * LS + sn * 16 + n_] = s[r] * scale;
        }
        __syncthreads();

        // Online softmax: 4 threads per row, 8 cols each.
        {
            const int row = tid >> 2;
            const int qq  = tid & 3;
            float sv[8];
            float mloc = -INFINITY;
            for (int c = 0; c < 8; ++c) {
                sv[c] = Sc[row * LS + qq * 8 + c];
                mloc = fmaxf(mloc, sv[c]);
            }
            mloc = fmaxf(mloc, __shfl_xor(mloc, 1, 32));
            mloc = fmaxf(mloc, __shfl_xor(mloc, 2, 32));
            float mold = mrow[row];
            float mnew = fmaxf(mold, mloc);
            float al   = __expf(mold - mnew);     // -inf -> 0 on first tile
            float ps   = 0.f;
            for (int c = 0; c < 8; ++c) {
                float p = __expf(sv[c] - mnew);
                ps += p;
                Ps[row * LP + qq * 8 + c] = (bf16)p;
            }
            ps += __shfl_xor(ps, 1, 32);
            ps += __shfl_xor(ps, 2, 32);
            if (qq == 0) {
                mrow[row] = mnew;
                lrow[row] = lrow[row] * al + ps;
                arow[row] = al;
            }
        }
        __syncthreads();

        // Rescale accumulators and add P * V (K-dim 32 = 1 chunk).
        {
            float al[8];
            for (int r = 0; r < 8; ++r) al[r] = arow[mi * 16 + r + 8 * hf];
            v16bf pa = load_frag_a(&Ps[(mi * 16) * LP], LP);
            for (int t = 0; t < 4; ++t) {
                int ni = nb + t;
                for (int r = 0; r < 8; ++r) accO[t][r] *= al[r];
                v16bf vb = load_frag_bt(&Vt[(ni * 16) * LV], LV);
                wmma_bf16(accO[t], pa, vb);
            }
        }
    }
    __syncthreads();

    // Final: divide by row sums, emit bf16 to [B, S, H*D].
    {
        float li[8];
        for (int r = 0; r < 8; ++r) li[r] = 1.f / lrow[mi * 16 + r + 8 * hf];
        for (int t = 0; t < 4; ++t) {
            int ni = nb + t;
            for (int r = 0; r < 8; ++r) {
                int srow = q0 + mi * 16 + r + 8 * hf;
                int col  = h * HD + ni * 16 + n_;
                og[(size_t)(b * SEQ + srow) * (NH * HD) + col] =
                    (bf16)(accO[t][r] * li[r]);
            }
        }
    }
}

// ---------------------------------------------------------------------------
// Launch: casts -> qkv GEMM -> rope/norm -> flash attention -> out GEMM
// ---------------------------------------------------------------------------
extern "C" void kernel_launch(void* const* d_in, const int* in_sizes, int n_in,
                              void* d_out, int out_size, void* d_ws, size_t ws_size,
                              hipStream_t stream) {
    (void)in_sizes; (void)n_in; (void)out_size; (void)ws_size;
    const float* hidden   = (const float*)d_in[0];
    const float* cosp     = (const float*)d_in[1];
    const float* sinp     = (const float*)d_in[2];
    const float* qkv_w    = (const float*)d_in[3];
    const float* o_w      = (const float*)d_in[4];
    const float* q_norm_w = (const float*)d_in[5];
    const float* k_norm_w = (const float*)d_in[6];
    float* out = (float*)d_out;

    char* ws = (char*)d_ws;
    size_t off = 0;
    float* qkv  = (float*)(ws + off); off += (size_t)NTOK * QKVN * 4;            // 100.7 MB
    bf16* qbf   = (bf16*)(ws + off);  off += (size_t)BATCH * NH  * SEQ * HD * 2; //  33.6 MB
    bf16* kbf   = (bf16*)(ws + off);  off += (size_t)BATCH * NKV * SEQ * HD * 2; //   8.4 MB
    bf16* vbf   = (bf16*)(ws + off);  off += (size_t)BATCH * NKV * SEQ * HD * 2; //   8.4 MB
    bf16* obf   = (bf16*)(ws + off);  off += (size_t)NTOK * HID * 2;             //  33.6 MB
    bf16* hbf   = (bf16*)(ws + off);  off += (size_t)NTOK * HID * 2;             //  33.6 MB
    bf16* wqkvb = (bf16*)(ws + off);  off += (size_t)QKVN * HID * 2;             //  50.3 MB
    bf16* wob   = (bf16*)(ws + off);                                             //  33.6 MB

    // 0) one-shot bf16 casts (amortized: ~250 MB traffic vs 100s of us compute)
    {
        int n8;
        n8 = (NTOK * HID) / 8;
        cast_f32_bf16<<<(n8 + 255) / 256, 256, 0, stream>>>(hidden, hbf, n8);
        n8 = (QKVN * HID) / 8;
        cast_f32_bf16<<<(n8 + 255) / 256, 256, 0, stream>>>(qkv_w, wqkvb, n8);
        n8 = (HID * HID) / 8;
        cast_f32_bf16<<<(n8 + 255) / 256, 256, 0, stream>>>(o_w, wob, n8);
    }
    // 1) qkv = hidden @ qkv_w^T   (M=4096, N=6144, K=4096)
    {
        dim3 grid(QKVN / 128, NTOK / 128);
        gemm_wmma<<<grid, 256, 0, stream>>>(hbf, wqkvb, qkv, NTOK, QKVN, HID);
    }
    // 2) RoPE + RMSNorm + cast, head-major layouts
    rope_norm_kernel<<<NTOK, 256, 0, stream>>>(qkv, cosp, sinp,
                                               q_norm_w, k_norm_w,
                                               qbf, kbf, vbf);
    // 3) attention
    {
        dim3 grid(SEQ / 64, BATCH * NH);
        flash_attn_kernel<<<grid, 256, 0, stream>>>(qbf, kbf, vbf, obf);
    }
    // 4) out = attn @ o_w^T   (M=4096, N=4096, K=4096)
    {
        dim3 grid(HID / 128, NTOK / 128);
        gemm_wmma<<<grid, 256, 0, stream>>>(obf, wob, out, NTOK, HID, HID);
    }
}
